// Attention_37022618091705
// MI455X (gfx1250) — compile-verified, matrix operand order checked by reading, not run on hardware
//
#include <hip/hip_runtime.h>
#include <hip/hip_bf16.h>
#include <math.h>

#define T_SEQ 2048
#define DM    2048
#define NH    16
#define DHEAD 128

typedef __attribute__((ext_vector_type(16))) __bf16 v16bf;
typedef __attribute__((ext_vector_type(8)))  float  v8f;
typedef __bf16 bf16_t;

// ---------- helpers ----------
__device__ __forceinline__ bf16_t f2bf(float f) {
  unsigned int u = __builtin_bit_cast(unsigned int, f);
  unsigned int r = (u + 0x7FFFu + ((u >> 16) & 1u)) >> 16;   // RNE
  return __builtin_bit_cast(bf16_t, (unsigned short)r);
}

// ISA 7.12.2: 16-bit A (16x32) / B (32x16) fragment K index for element e,
// lane-half g. K chunks of 8 are split between lane halves.
__device__ __forceinline__ int kmap(int e, int g) {
  return e + (g ? 8 : 0) + ((e >= 8) ? 8 : 0);
}

__device__ __forceinline__ v8f wmma_bf16(v16bf a, v16bf b, v8f c) {
  return __builtin_amdgcn_wmma_f32_16x16x32_bf16(
      /*neg_a=*/false, a, /*neg_b=*/false, b,
      /*c_mod=*/(short)0, c, /*reuse_a=*/false, /*reuse_b=*/false);
}

// Async global->LDS copy (CDNA5, tracked by ASYNCcnt).
// LDS address = low 32 bits of the generic pointer (aperture rule: ADDR[31:0]).
__device__ __forceinline__ void async_copy_b128(void* lds, const void* g) {
  unsigned l = (unsigned)(unsigned long long)(uintptr_t)lds;
  unsigned long long ga = (unsigned long long)(uintptr_t)g;
  asm volatile("global_load_async_to_lds_b128 %0, %1, off"
               :: "v"(l), "v"(ga) : "memory");
}

// ---------- RMSNorm -> bf16 ----------
__global__ __launch_bounds__(256) void rmsnorm_bf16_kernel(
    const float* __restrict__ x, const float* __restrict__ w,
    bf16_t* __restrict__ h) {
  int t = blockIdx.x;
  const float* row = x + (size_t)t * DM;
  int tid = threadIdx.x;
  float ss = 0.f;
  for (int i = tid; i < DM; i += 256) { float v = row[i]; ss += v * v; }
#pragma unroll
  for (int m = 16; m >= 1; m >>= 1) ss += __shfl_xor(ss, m, 32);
  __shared__ float red[8];
  __shared__ float sscale;
  if ((tid & 31) == 0) red[tid >> 5] = ss;
  __syncthreads();
  if (tid == 0) {
    float tot = 0.f;
    for (int i = 0; i < 8; ++i) tot += red[i];
    sscale = rsqrtf(tot / (float)DM + 1e-5f);
  }
  __syncthreads();
  float scl = sscale;
  for (int i = tid; i < DM; i += 256)
    h[(size_t)t * DM + i] = f2bf(w[i] * row[i] * scl);
}

// ---------- fp32 -> bf16 convert (keeps [N][K] orientation) ----------
__global__ __launch_bounds__(256) void convert_bf16_kernel(
    const float* __restrict__ W, bf16_t* __restrict__ Wb) {
  size_t i = ((size_t)blockIdx.x * 256 + threadIdx.x) * 4;
  float4 v = *(const float4*)&W[i];
  __align__(8) bf16_t tmp[4] = {f2bf(v.x), f2bf(v.y), f2bf(v.z), f2bf(v.w)};
  *(uint2*)&Wb[i] = *(const uint2*)tmp;
}

// ---------- C[MxN] = A[MxK](bf16) * Bn[NxK](bf16)^T (+residual), fp32 out ----
#define GTM 128
#define GTN 64
#define GTK 32
#define APAD 8     // +16B row pad: keeps b128 alignment, spreads banks

__global__ __launch_bounds__(256) void gemm_bf16_kernel(
    const bf16_t* __restrict__ A, const bf16_t* __restrict__ Bn,
    const float* __restrict__ residual, float* __restrict__ C,
    int M, int N, int K) {
  __shared__ __align__(16) bf16_t sA[2][GTM][GTK + APAD];   // 20 KB
  __shared__ __align__(16) bf16_t sB[2][GTN][GTK + APAD];   // 10 KB
  int tid = threadIdx.x;
  int wave = tid >> 5, lane = tid & 31;
  int g = lane >> 4, la = lane & 15;
  int bm = blockIdx.x * GTM, bn = blockIdx.y * GTN;
  int wm = (wave >> 1) * 32, wn = (wave & 1) * 32;   // 4x2 wave grid, 32x32 each

  v8f acc[2][2] = {};
  int nk = K / GTK;

  auto stage = [&](int buf, int k0) {
#pragma unroll
    for (int j = 0; j < 2; ++j) {            // A tile: 512 x 16B chunks
      int c = tid * 2 + j;
      int row = c >> 2, col = (c & 3) * 8;
      *(uint4*)&sA[buf][row][col] =
          *(const uint4*)&A[(size_t)(bm + row) * K + k0 + col];
    }
    {                                        // B tile: [n][k], 256 x 16B chunks
      int row = tid >> 2, col = (tid & 3) * 8;
      *(uint4*)&sB[buf][row][col] =
          *(const uint4*)&Bn[(size_t)(bn + row) * K + k0 + col];
    }
  };

  stage(0, 0);
  __syncthreads();
  for (int ks = 0; ks < nk; ++ks) {
    int buf = ks & 1;
    if (ks + 1 < nk) {
      __builtin_prefetch(&A[(size_t)bm * K + (size_t)(ks + 1) * GTK], 0, 1);
      stage(buf ^ 1, (ks + 1) * GTK);
    }
    v16bf a0, a1, b0, b1;
#pragma unroll
    for (int e = 0; e < 16; ++e) {           // all K-contiguous -> ds_load_b128
      int kk = kmap(e, g);
      a0[e] = sA[buf][wm + la][kk];
      a1[e] = sA[buf][wm + 16 + la][kk];
      b0[e] = sB[buf][wn + la][kk];
      b1[e] = sB[buf][wn + 16 + la][kk];
    }
    acc[0][0] = wmma_bf16(a0, b0, acc[0][0]);
    acc[0][1] = wmma_bf16(a0, b1, acc[0][1]);
    acc[1][0] = wmma_bf16(a1, b0, acc[1][0]);
    acc[1][1] = wmma_bf16(a1, b1, acc[1][1]);
    __syncthreads();
  }

#pragma unroll
  for (int i = 0; i < 2; ++i)
#pragma unroll
    for (int j = 0; j < 2; ++j)
#pragma unroll
      for (int r = 0; r < 8; ++r) {
        int m = bm + wm + i * 16 + r + (g ? 8 : 0);
        int n = bn + wn + j * 16 + la;
        float v = acc[i][j][r];
        if (residual) v += residual[(size_t)m * N + n];
        C[(size_t)m * N + n] = v;
      }
}

// ---------- RoPE + convert ----------
// mode 1: rope, dst layout [H][T][128]   (Q, K)
// mode 0: no rope, dst layout [H][128][T] (V, transposed for WMMA B tiles)
__global__ __launch_bounds__(256) void rope_convert_kernel(
    const float* __restrict__ src, const float* __restrict__ cosT,
    const float* __restrict__ sinT, bf16_t* __restrict__ dst, int mode) {
  int t = blockIdx.x;
  for (int i = threadIdx.x; i < DM; i += 256) {
    int h = i >> 7, d = i & 127;
    float v = src[(size_t)t * DM + i];
    if (mode == 1) {
      float c = cosT[t * DHEAD + d], s = sinT[t * DHEAD + d];
      float other = src[(size_t)t * DM + h * DHEAD + ((d < 64) ? d + 64 : d - 64)];
      float r = (d < 64) ? (v * c - other * s) : (v * c + other * s);
      dst[((size_t)h * T_SEQ + t) * DHEAD + d] = f2bf(r);
    } else {
      dst[((size_t)h * DHEAD + d) * T_SEQ + t] = f2bf(v);
    }
  }
}

// ---------- causal flash attention, WMMA QK^T and PV ----------
__global__ __launch_bounds__(256) void flash_attn_kernel(
    const bf16_t* __restrict__ Qh, const bf16_t* __restrict__ Kh,
    const bf16_t* __restrict__ Vt, bf16_t* __restrict__ OutTD) {
  __shared__ __align__(16) bf16_t sK[32][DHEAD + APAD];   // 8.5 KB, [krow][d]
  __shared__ __align__(16) bf16_t sVt[DHEAD][32 + APAD];  // 10 KB, [d][krow]
  __shared__ __align__(16) bf16_t sP[8][16][32 + APAD];   // 10 KB, per-wave P
  int h = blockIdx.x, qblk = blockIdx.y;
  int tid = threadIdx.x, wv = tid >> 5, lane = tid & 31;
  int g = lane >> 4, la = lane & 15;
  int qb = qblk * 128 + wv * 16;                   // this wave's 16 q rows
  const size_t hb = (size_t)h * T_SEQ;

  // Q fragments: 16 rows x 128 d, as 4 K-chunks of 32
  v16bf qf[4];
#pragma unroll
  for (int dc = 0; dc < 4; ++dc)
#pragma unroll
    for (int e = 0; e < 16; ++e)
      qf[dc][e] = Qh[(hb + qb + la) * DHEAD + dc * 32 + kmap(e, g)];

  v8f o[8] = {};
  float Mr[8], Lr[8];
#pragma unroll
  for (int r = 0; r < 8; ++r) { Mr[r] = -3.0e38f; Lr[r] = 0.f; }

  int ktiles = qblk * 4 + 4;   // causal extent of this 128-row q block
  for (int kt = 0; kt < ktiles; ++kt) {
    int k0 = kt * 32;
    // stage K (32x128, row-major) and V^T (128x32) via async global->LDS
#pragma unroll
    for (int j = 0; j < 2; ++j) {
      int c = tid * 2 + j;
      {  // K: row = k-row, col = d
        int row = c >> 4, col = (c & 15) * 8;
        async_copy_b128(&sK[row][col], &Kh[(hb + k0 + row) * DHEAD + col]);
      }
      {  // V^T: row = d, col = k-row
        int row = c >> 2, col = (c & 3) * 8;
        async_copy_b128(&sVt[row][col],
                        &Vt[((size_t)h * DHEAD + row) * T_SEQ + k0 + col]);
      }
    }
    asm volatile("s_wait_asynccnt 0x0" ::: "memory");
    __syncthreads();
    if (k0 <= qb + 15) {                 // wave-uniform: EXEC stays all-ones
      // S(16x32) = Q(16x128) . K^T : B(d,n) = K[k0+n][d]
      v8f s0 = {}, s1 = {};
#pragma unroll
      for (int dc = 0; dc < 4; ++dc) {
        v16bf b0, b1;
#pragma unroll
        for (int e = 0; e < 16; ++e) {
          int kk = dc * 32 + kmap(e, g);
          b0[e] = sK[la][kk];
          b1[e] = sK[16 + la][kk];
        }
        s0 = wmma_bf16(qf[dc], b0, s0);
        s1 = wmma_bf16(qf[dc], b1, s1);
      }
      const float sc = 0.08838834764831845f;  // 1/sqrt(128)
#pragma unroll
      for (int r = 0; r < 8; ++r) {
        int rowg = qb + r + (g ? 8 : 0);
        float v0 = s0[r] * sc, v1 = s1[r] * sc;
        if (k0 + la > rowg)      v0 = -3.0e38f;   // causal mask
        if (k0 + 16 + la > rowg) v1 = -3.0e38f;
        float m = fmaxf(v0, v1);
#pragma unroll
        for (int sh = 1; sh < 16; sh <<= 1) m = fmaxf(m, __shfl_xor(m, sh, 32));
        float mnew = fmaxf(Mr[r], m);
        float corr = __expf(Mr[r] - mnew);
        Mr[r] = mnew;
        float p0 = __expf(v0 - mnew);
        float p1 = __expf(v1 - mnew);
        s0[r] = p0; s1[r] = p1;
        float rs = p0 + p1;
#pragma unroll
        for (int sh = 1; sh < 16; sh <<= 1) rs += __shfl_xor(rs, sh, 32);
        Lr[r] = Lr[r] * corr + rs;
#pragma unroll
        for (int dt = 0; dt < 8; ++dt) o[dt][r] *= corr;
      }
      // relay P through per-wave LDS to repack C-layout -> A-layout
#pragma unroll
      for (int r = 0; r < 8; ++r) {
        sP[wv][r + (g ? 8 : 0)][la]      = f2bf(s0[r]);
        sP[wv][r + (g ? 8 : 0)][16 + la] = f2bf(s1[r]);
      }
      asm volatile("s_wait_dscnt 0x0" ::: "memory");   // cross-lane LDS in-wave
      v16bf pa;
#pragma unroll
      for (int e = 0; e < 16; ++e) pa[e] = sP[wv][la][kmap(e, g)];
      // O(16x128) += P(16x32) . V(32x128); B(k,d) = V^T[d][k], K-contiguous
#pragma unroll
      for (int dt = 0; dt < 8; ++dt) {
        v16bf bv;
#pragma unroll
        for (int e = 0; e < 16; ++e) bv[e] = sVt[dt * 16 + la][kmap(e, g)];
        o[dt] = wmma_bf16(pa, bv, o[dt]);
      }
    }
    __syncthreads();
  }

  // normalize, write back as bf16 [T][DM] (head-concat) for the wo GEMM
#pragma unroll
  for (int dt = 0; dt < 8; ++dt)
#pragma unroll
    for (int r = 0; r < 8; ++r) {
      int trow = qb + r + (g ? 8 : 0);
      OutTD[(size_t)trow * DM + h * DHEAD + dt * 16 + la] = f2bf(o[dt][r] / Lr[r]);
    }
}

// ---------- launch ----------
extern "C" void kernel_launch(void* const* d_in, const int* in_sizes, int n_in,
                              void* d_out, int out_size, void* d_ws, size_t ws_size,
                              hipStream_t stream) {
  (void)in_sizes; (void)n_in; (void)out_size; (void)ws_size;
  const float* x    = (const float*)d_in[0];
  const float* w_ln = (const float*)d_in[1];
  const float* wq   = (const float*)d_in[2];
  const float* wk   = (const float*)d_in[3];
  const float* wv   = (const float*)d_in[4];
  const float* wo   = (const float*)d_in[5];
  const float* cosT = (const float*)d_in[6];
  const float* sinT = (const float*)d_in[7];
  float* out = (float*)d_out;

  const size_t TD = (size_t)T_SEQ * DM;
  char* p = (char*)d_ws;                       // ~80 MB total
  bf16_t* h_bf  = (bf16_t*)p;  p += TD * 2;
  bf16_t* wqb   = (bf16_t*)p;  p += TD * 2;
  bf16_t* wkb   = (bf16_t*)p;  p += TD * 2;
  bf16_t* wvb   = (bf16_t*)p;  p += TD * 2;
  bf16_t* wob   = (bf16_t*)p;  p += TD * 2;
  float*  tmp_f = (float*)p;   p += TD * 4;    // reused for q/k/v fp32 projections
  bf16_t* qh    = (bf16_t*)p;  p += TD * 2;
  bf16_t* kh    = (bf16_t*)p;  p += TD * 2;
  bf16_t* vt    = (bf16_t*)p;  p += TD * 2;
  bf16_t* attn  = (bf16_t*)tmp_f;              // reuse: attention runs after ropes

  dim3 blk(256);
  rmsnorm_bf16_kernel<<<T_SEQ, blk, 0, stream>>>(x, w_ln, h_bf);

  int cg = (int)(TD / (256 * 4));
  convert_bf16_kernel<<<cg, blk, 0, stream>>>(wq, wqb);
  convert_bf16_kernel<<<cg, blk, 0, stream>>>(wk, wkb);
  convert_bf16_kernel<<<cg, blk, 0, stream>>>(wv, wvb);
  convert_bf16_kernel<<<cg, blk, 0, stream>>>(wo, wob);

  dim3 gg(T_SEQ / GTM, DM / GTN);
  gemm_bf16_kernel<<<gg, blk, 0, stream>>>(h_bf, wqb, nullptr, tmp_f, T_SEQ, DM, DM);
  rope_convert_kernel<<<T_SEQ, blk, 0, stream>>>(tmp_f, cosT, sinT, qh, 1);
  gemm_bf16_kernel<<<gg, blk, 0, stream>>>(h_bf, wkb, nullptr, tmp_f, T_SEQ, DM, DM);
  rope_convert_kernel<<<T_SEQ, blk, 0, stream>>>(tmp_f, cosT, sinT, kh, 1);
  gemm_bf16_kernel<<<gg, blk, 0, stream>>>(h_bf, wvb, nullptr, tmp_f, T_SEQ, DM, DM);
  rope_convert_kernel<<<T_SEQ, blk, 0, stream>>>(tmp_f, cosT, sinT, vt, 0);

  dim3 ag(NH, T_SEQ / 128);
  flash_attn_kernel<<<ag, blk, 0, stream>>>(qh, kh, vt, attn);

  gemm_bf16_kernel<<<gg, blk, 0, stream>>>(attn, wob, x, out, T_SEQ, DM, DM);
}
